// GraphMambaConv_23167053595296
// MI455X (gfx1250) — compile-verified, matrix operand order checked by reading, not run on hardware
//
#include <hip/hip_runtime.h>
#include <hip/hip_bf16.h>
#include <math.h>

// ---------------- sizes ----------------
#define NN 32768
#define CC 128
#define GG 64
#define LL 512
#define EE 524288
#define DI 256           // D_INNER
#define DS 16            // D_STATE
#define DTR 8            // DT_RANK
#define PROJW 48         // padded 40 -> 48
#define EPS 1e-5f
#define WAVES_PER_BLOCK 8
#define MTILES (NN / 16)             // 2048
#define GEMM_GRID_Y (MTILES / WAVES_PER_BLOCK)  // 256

typedef __attribute__((ext_vector_type(16))) __bf16 v16bf;
typedef __attribute__((ext_vector_type(8)))  float  v8f;
typedef __attribute__((ext_vector_type(4)))  float  v4f;

__device__ __forceinline__ __bf16 to_bf16(float f) {
    unsigned int u = __float_as_uint(f);
    unsigned int r = u + 0x7FFFu + ((u >> 16) & 1u);   // round-to-nearest-even
    unsigned short h = (unsigned short)(r >> 16);
    return __builtin_bit_cast(__bf16, h);
}

// ---------------- utility kernels ----------------
__global__ void k_zero(float* p, size_t n) {
    size_t i = (size_t)blockIdx.x * blockDim.x + threadIdx.x;
    size_t stride = (size_t)gridDim.x * blockDim.x;
    for (; i < n; i += stride) p[i] = 0.f;
}

// segment_sum(x[src] -> agg[dst]) with f32 atomics; 32 lanes x 4 ch per edge.
__global__ void k_scatter_add(const float* __restrict__ x, const int* __restrict__ ei,
                              float* __restrict__ agg) {
    int tid  = blockIdx.x * blockDim.x + threadIdx.x;
    int e    = tid >> 5;
    int lane = tid & 31;
    if (e >= EE) return;
    int src = ei[e];
    int dst = ei[EE + e];
    const float* xs = x   + (size_t)src * CC + lane * 4;
    float*       ad = agg + (size_t)dst * CC + lane * 4;
#pragma unroll
    for (int j = 0; j < 4; ++j) atomicAdd(ad + j, xs[j]);
}

// ---------------- operand packing into WMMA bf16 fragment layout ----------------
// Fragment layout (ISA 7.12.2, 16-bit A 16x32): lane holds 16 bf16; element e maps to
// k = kb + (e<8 ? e : e+8), kb = kc*32 + (lane>=16 ? 8 : 0); row/col = lane&15.
// Packed storage: frag[((tile*KCtot + kc0+kc)*32 + lane)*16 + e]

// A (activations, row-major [M,K], K multiple of 32), optional per-k affine (folded BN)
template <bool AFF>
__global__ __launch_bounds__(256) void k_pack_a(
    const float* __restrict__ A, const float* __restrict__ sc, const float* __restrict__ sh,
    __bf16* __restrict__ out, int K, int KC, int KCtot, int kc0)
{
    int tid   = blockIdx.x * 256 + threadIdx.x;
    int lane  = tid & 31;
    int rest  = tid >> 5;
    int kc    = rest % KC;
    int mtile = rest / KC;
    if (mtile >= MTILES) return;
    const int m  = mtile * 16 + (lane & 15);
    const int kb = kc * 32 + ((lane >> 4) << 3);
    const float* ap = A + (size_t)m * K + kb;
    v4f q0 = *(const v4f*)(ap);
    v4f q1 = *(const v4f*)(ap + 4);
    v4f q2 = *(const v4f*)(ap + 16);
    v4f q3 = *(const v4f*)(ap + 20);
    __bf16* o = out + (((size_t)mtile * KCtot + kc0 + kc) * 32 + lane) * 16;
#pragma unroll
    for (int e = 0; e < 4; ++e) {
        float v0 = q0[e], v1 = q1[e], v2 = q2[e], v3 = q3[e];
        if (AFF) {
            v0 = v0 * sc[kb + e]      + sh[kb + e];
            v1 = v1 * sc[kb + 4 + e]  + sh[kb + 4 + e];
            v2 = v2 * sc[kb + 16 + e] + sh[kb + 16 + e];
            v3 = v3 * sc[kb + 20 + e] + sh[kb + 20 + e];
        }
        o[e]      = to_bf16(v0);
        o[4 + e]  = to_bf16(v1);
        o[8 + e]  = to_bf16(v2);
        o[12 + e] = to_bf16(v3);
    }
}

// B (weights, row-major [K,Nreal]); zero-pads columns >= Nreal
__global__ __launch_bounds__(256) void k_pack_b(
    const float* __restrict__ W, __bf16* __restrict__ out,
    int Nreal, int Ntiles, int KC, int KCtot, int kc0)
{
    int tid   = blockIdx.x * 256 + threadIdx.x;
    int lane  = tid & 31;
    int rest  = tid >> 5;
    int kc    = rest % KC;
    int ntile = rest / KC;
    if (ntile >= Ntiles) return;
    const int n  = ntile * 16 + (lane & 15);
    const int kb = kc * 32 + ((lane >> 4) << 3);
    __bf16* o = out + (((size_t)ntile * KCtot + kc0 + kc) * 32 + lane) * 16;
#pragma unroll
    for (int e = 0; e < 16; ++e) {
        const int k = kb + (e < 8 ? e : e + 8);
        float v = (n < Nreal) ? W[(size_t)k * Nreal + n] : 0.f;
        o[e] = to_bf16(v);
    }
}

// ---------------- WMMA GEMM on packed fragments ----------------
// Out[M,Nstride] = epilogue( apack @ bpack ); one 16x16 tile per wave, K = KC*32.
template <int KC, int ACT, bool BIAS, bool RESID>
__global__ __launch_bounds__(256) void k_gemm(
    const __bf16* __restrict__ apack, const __bf16* __restrict__ bpack,
    const float* __restrict__ bias, const float* __restrict__ resid,
    float* __restrict__ Out, int Nstride)
{
    const int lane  = threadIdx.x & 31;
    const int wave  = threadIdx.x >> 5;
    const int tileM = blockIdx.y * WAVES_PER_BLOCK + wave;
    const int tileN = blockIdx.x;

    const v16bf* ap = (const v16bf*)(apack + (((size_t)tileM * KC) * 32 + lane) * 16);
    const v16bf* bp = (const v16bf*)(bpack + (((size_t)tileN * KC) * 32 + lane) * 16);

    v8f acc = {};
#pragma unroll
    for (int kc = 0; kc < KC; ++kc) {
        v16bf a = ap[kc * 32];       // 32B packed fragment -> b128 pair
        v16bf b = bp[kc * 32];
        acc = __builtin_amdgcn_wmma_f32_16x16x32_bf16(
                  false, a, false, b, (short)0, acc, false, false);
    }

    const int nB    = tileN * 16 + (lane & 15);
    const int mbase = tileM * 16 + ((lane >> 4) << 3);
    const float bval = BIAS ? bias[nB] : 0.f;
#pragma unroll
    for (int r = 0; r < 8; ++r) {
        const int m = mbase + r;
        float v = acc[r] + bval;
        if (ACT == 1) v = 0.5f * v * (1.f + erff(v * 0.70710678118654752f)); // exact GELU
        if (RESID) v += resid[(size_t)m * Nstride + nB];
        Out[(size_t)m * Nstride + nB] = v;
    }
}

// ---------------- depthwise causal conv (d_conv=4) + SiLU ----------------
__global__ __launch_bounds__(256) void k_conv_silu(
    const float* __restrict__ xz, const float* __restrict__ conv_w,
    const float* __restrict__ conv_b, float* __restrict__ xc)
{
    const int n = blockIdx.x;        // global row (g*L + l)
    const int d = threadIdx.x;       // 0..255
    const int l = n & (LL - 1);
    float acc = conv_b[d];
#pragma unroll
    for (int k = 0; k < 4; ++k) {
        const int lk = l + k - 3;
        if (lk >= 0) acc += xz[(size_t)(n + k - 3) * (2 * DI) + d] * conv_w[d * 4 + k];
    }
    const float sig = 1.f / (1.f + __expf(-acc));
    xc[(size_t)n * DI + d] = acc * sig;
}

// ---------------- dt = softplus(dt_raw @ dt_W + dt_b), K = 8 (scalar) ----------------
__global__ __launch_bounds__(256) void k_dt(
    const float* __restrict__ proj, const float* __restrict__ dt_W,
    const float* __restrict__ dt_b, float* __restrict__ dt)
{
    const int n = blockIdx.x;
    const int d = threadIdx.x;
    float acc = dt_b[d];
#pragma unroll
    for (int r = 0; r < DTR; ++r)
        acc += proj[(size_t)n * PROJW + r] * dt_W[r * DI + d];
    const float sp = fmaxf(acc, 0.f) + log1pf(__expf(-fabsf(acc)));  // stable softplus
    dt[(size_t)n * DI + d] = sp;
}

// ---------------- selective scan: one block per graph, thread d keeps 16 states ----------
__global__ __launch_bounds__(256) void k_scan(
    const float* __restrict__ dt, const float* __restrict__ proj,
    const float* __restrict__ xc, const float* __restrict__ xz,
    const float* __restrict__ A_log, const float* __restrict__ Dp,
    float* __restrict__ y)
{
    const int g = blockIdx.x;
    const int d = threadIdx.x;
    float A[DS], h[DS];
#pragma unroll
    for (int s = 0; s < DS; ++s) {
        A[s] = -__expf(A_log[d * DS + s]);
        h[s] = 0.f;
    }
    const float Dd = Dp[d];
    __shared__ float Bs[DS], Cs[DS];

    for (int l = 0; l < LL; ++l) {
        const size_t n = (size_t)g * LL + l;
        if (d < 32) {                       // proj cols 8..23 = B, 24..39 = C
            const float v = proj[n * PROJW + 8 + d];
            if (d < 16) Bs[d] = v; else Cs[d - 16] = v;
        }
        __syncthreads();

        const float dtv = dt[n * DI + d];
        const float xv  = xc[n * DI + d];
        const float dx  = dtv * xv;
        float acc = 0.f;
#pragma unroll
        for (int s = 0; s < DS; ++s) {
            h[s] = h[s] * __expf(dtv * A[s]) + dx * Bs[s];
            acc += h[s] * Cs[s];
        }
        const float zv  = xz[n * (2 * DI) + DI + d];
        const float sig = 1.f / (1.f + __expf(-zv));
        y[n * DI + d] = (acc + xv * Dd) * (zv * sig);
        __syncthreads();
    }
}

// ---------------- BatchNorm stats: per column -> scale/shift (folds gamma/beta) -------
__global__ __launch_bounds__(256) void k_bn_stats(
    const float* __restrict__ X, int rows, int cols,
    const float* __restrict__ gamma, const float* __restrict__ beta,
    float* __restrict__ s_out, float* __restrict__ t_out)
{
    const int c = blockIdx.x;
    float sum = 0.f, sq = 0.f;
    for (int r = threadIdx.x; r < rows; r += blockDim.x) {
        const float v = X[(size_t)r * cols + c];
        sum += v; sq += v * v;
    }
    __shared__ float ssum[256], ssq[256];
    ssum[threadIdx.x] = sum; ssq[threadIdx.x] = sq;
    __syncthreads();
    for (int st = 128; st > 0; st >>= 1) {
        if (threadIdx.x < st) {
            ssum[threadIdx.x] += ssum[threadIdx.x + st];
            ssq [threadIdx.x] += ssq [threadIdx.x + st];
        }
        __syncthreads();
    }
    if (threadIdx.x == 0) {
        const float m   = ssum[0] / rows;
        const float var = ssq[0] / rows - m * m;
        const float sc  = gamma[c] * rsqrtf(var + EPS);
        s_out[c] = sc;
        t_out[c] = beta[c] - m * sc;
    }
}

// out0 = bn1(h1) + bn2(h2pre), elementwise with folded affines
__global__ void k_add_norm(const float* __restrict__ h1, const float* __restrict__ h2,
                           const float* __restrict__ s1, const float* __restrict__ t1,
                           const float* __restrict__ s2, const float* __restrict__ t2,
                           float* __restrict__ out0) {
    const size_t i = (size_t)blockIdx.x * blockDim.x + threadIdx.x;
    if (i >= (size_t)NN * CC) return;
    const int c = (int)(i & (CC - 1));
    out0[i] = s1[c] * h1[i] + t1[c] + s2[c] * h2[i] + t2[c];
}

// d_out = s[c]*x + t[c]
__global__ void k_affine(const float* __restrict__ x,
                         const float* __restrict__ s, const float* __restrict__ t,
                         float* __restrict__ out) {
    const size_t i = (size_t)blockIdx.x * blockDim.x + threadIdx.x;
    if (i >= (size_t)NN * CC) return;
    const int c = (int)(i & (CC - 1));
    out[i] = s[c] * x[i] + t[c];
}

// ---------------- orchestration ----------------
extern "C" void kernel_launch(void* const* d_in, const int* in_sizes, int n_in,
                              void* d_out, int out_size, void* d_ws, size_t ws_size,
                              hipStream_t stream) {
    const float* x        = (const float*)d_in[0];
    const int*   ei       = (const int*)  d_in[1];
    // d_in[2] = batch (unused: arange(N)//L == dense reshape)
    const float* W_root   = (const float*)d_in[3];
    const float* W_rel    = (const float*)d_in[4];
    const float* b_rel    = (const float*)d_in[5];
    const float* n1_g     = (const float*)d_in[6];
    const float* n1_b     = (const float*)d_in[7];
    const float* n2_g     = (const float*)d_in[8];
    const float* n2_b     = (const float*)d_in[9];
    const float* n3_g     = (const float*)d_in[10];
    const float* n3_b     = (const float*)d_in[11];
    const float* mbn1_g   = (const float*)d_in[12];
    const float* mbn1_b   = (const float*)d_in[13];
    const float* mlp_W1   = (const float*)d_in[14];
    const float* mlp_b1   = (const float*)d_in[15];
    const float* mbn2_g   = (const float*)d_in[16];
    const float* mbn2_b   = (const float*)d_in[17];
    const float* mlp_W2   = (const float*)d_in[18];
    const float* mlp_b2   = (const float*)d_in[19];
    const float* in_W     = (const float*)d_in[20];
    const float* conv_w   = (const float*)d_in[21];
    const float* conv_b   = (const float*)d_in[22];
    const float* xproj_W  = (const float*)d_in[23];
    const float* dt_W     = (const float*)d_in[24];
    const float* dt_b     = (const float*)d_in[25];
    const float* A_log    = (const float*)d_in[26];
    const float* Dp       = (const float*)d_in[27];
    const float* out_W    = (const float*)d_in[28];
    float* out = (float*)d_out;

    // workspace layout (floats; all offsets multiples of 256 -> 32B aligned)
    float* ws = (float*)d_ws;
    size_t off = 0;
    auto alloc = [&](size_t n) { float* p = ws + off; off += n; return p; };
    float* agg   = alloc((size_t)NN * CC);
    float* h1    = alloc((size_t)NN * CC);
    float* xz    = alloc((size_t)NN * 2 * DI);
    float* xc    = alloc((size_t)NN * DI);
    float* proj  = alloc((size_t)NN * PROJW);
    float* dtb   = alloc((size_t)NN * DI);
    float* ybuf  = alloc((size_t)NN * DI);
    float* h2pre = alloc((size_t)NN * CC);
    float* out0  = alloc((size_t)NN * CC);
    float* m1g   = alloc((size_t)NN * 2 * CC);
    float* out1  = alloc((size_t)NN * CC);
    __bf16* apack = (__bf16*)alloc((size_t)NN * DI / 2);   // M x 256 bf16 max
    __bf16* bpack = (__bf16*)alloc(131072 / 2);            // 128K bf16 max
    float* s1 = alloc(256);  float* t1 = alloc(256);
    float* s2 = alloc(256);  float* t2 = alloc(256);
    float* s4 = alloc(256);  float* t4 = alloc(256);
    float* s5 = alloc(256);  float* t5 = alloc(256);
    float* s6 = alloc(256);  float* t6 = alloc(256);
    (void)ws_size; (void)n_in; (void)in_sizes; (void)out_size;

    const dim3 blk(256);

    auto packA = [&](const float* A, int K, int KC, int KCtot, int kc0,
                     const float* s, const float* t) {
        const int nthr = MTILES * KC * 32;
        if (s) k_pack_a<true ><<<nthr / 256, blk, 0, stream>>>(A, s, t, apack, K, KC, KCtot, kc0);
        else   k_pack_a<false><<<nthr / 256, blk, 0, stream>>>(A, nullptr, nullptr, apack, K, KC, KCtot, kc0);
    };
    auto packB = [&](const float* W, int Nreal, int Ntiles, int KC, int KCtot, int kc0) {
        const int nthr = Ntiles * KC * 32;
        k_pack_b<<<(nthr + 255) / 256, blk, 0, stream>>>(W, bpack, Nreal, Ntiles, KC, KCtot, kc0);
    };

    // 1) GraphConv branch: agg = segment_sum(x[src] -> dst)
    k_zero<<<2048, blk, 0, stream>>>(agg, (size_t)NN * CC);
    k_scatter_add<<<(EE * 32) / 256, blk, 0, stream>>>(x, ei, agg);
    // h1 = agg@W_rel + x@W_root + b_rel + x  (K-concat: [agg|x] @ [W_rel;W_root])
    packA(agg, CC, 4, 8, 0, nullptr, nullptr);
    packA(x,   CC, 4, 8, 4, nullptr, nullptr);
    packB(W_rel,  CC, CC / 16, 4, 8, 0);
    packB(W_root, CC, CC / 16, 4, 8, 4);
    k_gemm<8, 0, true, true><<<dim3(CC / 16, GEMM_GRID_Y), blk, 0, stream>>>(
        apack, bpack, b_rel, x, h1, CC);

    // 2) Mamba branch
    // xz = x @ in_W   (N=512, K=128)
    packA(x, CC, 4, 4, 0, nullptr, nullptr);
    packB(in_W, 2 * DI, 2 * DI / 16, 4, 4, 0);
    k_gemm<4, 0, false, false><<<dim3(2 * DI / 16, GEMM_GRID_Y), blk, 0, stream>>>(
        apack, bpack, nullptr, nullptr, xz, 2 * DI);
    k_conv_silu<<<NN, blk, 0, stream>>>(xz, conv_w, conv_b, xc);
    // proj = xc @ xproj_W  (Nreal=40 zero-padded to 48, K=256)
    packA(xc, DI, 8, 8, 0, nullptr, nullptr);
    packB(xproj_W, DTR + 2 * DS, PROJW / 16, 8, 8, 0);
    k_gemm<8, 0, false, false><<<dim3(PROJW / 16, GEMM_GRID_Y), blk, 0, stream>>>(
        apack, bpack, nullptr, nullptr, proj, PROJW);
    k_dt<<<NN, blk, 0, stream>>>(proj, dt_W, dt_b, dtb);
    k_scan<<<GG, blk, 0, stream>>>(dtb, proj, xc, xz, A_log, Dp, ybuf);
    // h2pre = y @ out_W + x   (K=256)
    packA(ybuf, DI, 8, 8, 0, nullptr, nullptr);
    packB(out_W, CC, CC / 16, 8, 8, 0);
    k_gemm<8, 0, false, true><<<dim3(CC / 16, GEMM_GRID_Y), blk, 0, stream>>>(
        apack, bpack, nullptr, x, h2pre, CC);

    // 3) out0 = bn1(h1) + bn2(h2pre)
    k_bn_stats<<<CC, blk, 0, stream>>>(h1, NN, CC, n1_g, n1_b, s1, t1);
    k_bn_stats<<<CC, blk, 0, stream>>>(h2pre, NN, CC, n2_g, n2_b, s2, t2);
    k_add_norm<<<(NN * CC) / 256, blk, 0, stream>>>(h1, h2pre, s1, t1, s2, t2, out0);

    // 4) gated MLP (BN folded into A-pack affine)
    k_bn_stats<<<CC, blk, 0, stream>>>(out0, NN, CC, mbn1_g, mbn1_b, s4, t4);
    // m1g = gelu(bn(out0) @ mlp_W1 + b1)   (K=128, N=256)
    packA(out0, CC, 4, 4, 0, s4, t4);
    packB(mlp_W1, 2 * CC, 2 * CC / 16, 4, 4, 0);
    k_gemm<4, 1, true, false><<<dim3(2 * CC / 16, GEMM_GRID_Y), blk, 0, stream>>>(
        apack, bpack, mlp_b1, nullptr, m1g, 2 * CC);
    k_bn_stats<<<2 * CC, blk, 0, stream>>>(m1g, NN, 2 * CC, mbn2_g, mbn2_b, s5, t5);
    // out1 = out0 + (bn(m1g) @ mlp_W2 + b2)   (K=256, N=128)
    packA(m1g, 2 * CC, 8, 8, 0, s5, t5);
    packB(mlp_W2, CC, CC / 16, 8, 8, 0);
    k_gemm<8, 0, true, true><<<dim3(CC / 16, GEMM_GRID_Y), blk, 0, stream>>>(
        apack, bpack, mlp_b2, out0, out1, CC);

    // 5) final bn3
    k_bn_stats<<<CC, blk, 0, stream>>>(out1, NN, CC, n3_g, n3_b, s6, t6);
    k_affine<<<(NN * CC) / 256, blk, 0, stream>>>(out1, s6, t6, out);
}
